// KimiMoE_10746008175015
// MI455X (gfx1250) — compile-verified, matrix operand order checked by reading, not run on hardware
//
#include <hip/hip_runtime.h>
#include <hip/hip_bf16.h>
#include <math.h>

// ---------------- problem constants (match reference) ----------------
constexpr int TN  = 4096;          // tokens
constexpr int HN  = 1024;          // hidden
constexpr int IN  = 512;           // moe intermediate
constexpr int EN  = 64;            // experts
constexpr int KN  = 8;             // top-k
constexpr int NGN = 8;             // expert groups
constexpr int TKG = 4;             // top-k groups
constexpr int ISN = 1024;          // shared intermediate (I*NSH)
constexpr float RSCALE = 2.5f;

// ---------------- WMMA types ----------------
typedef __attribute__((ext_vector_type(16))) __bf16 bf16x16;
typedef __attribute__((ext_vector_type(8)))  float  floatx8;

union FragU { bf16x16 v; unsigned int u[8]; };

__device__ __forceinline__ unsigned short f2bf(float f) {
  union { float f; unsigned int u; } cv; cv.f = f;
  unsigned int u = cv.u;
  u += 0x7FFFu + ((u >> 16) & 1u);     // round-to-nearest-even
  return (unsigned short)(u >> 16);
}

// ---------------- async global->LDS staging (CDNA5 ASYNCcnt path) ----
#define USE_ASYNC_COPY 1

__device__ __forceinline__ void async_a_b128(const void* gsrc, void* ldst) {
#if USE_ASYNC_COPY
  // Generic LDS pointer: low 32 bits are the LDS byte address (aperture rule).
  unsigned l = (unsigned)(unsigned long long)ldst;
  asm volatile("global_load_async_to_lds_b128 %0, %1, off"
               :: "v"(l), "v"(gsrc) : "memory");
#else
  *(uint4*)ldst = *(const uint4*)gsrc;
#endif
}
__device__ __forceinline__ void async_wait() {
#if USE_ASYNC_COPY
  asm volatile("s_wait_asynccnt 0" ::: "memory");
#endif
}

// Fragment load from LDS (row-major ushort, stride ldk).
// lane row = <row>, half = lane>>4 selects K chunks {0..7,16..23} / {8..15,24..31}
__device__ __forceinline__ bf16x16 load_frag(const unsigned short* lds, int ldk,
                                             int row, int half) {
  const unsigned short* p = lds + row * ldk + half * 8;
  FragU f;
#pragma unroll
  for (int j = 0; j < 4; ++j) f.u[j]     = *(const unsigned int*)(p + 2 * j);
#pragma unroll
  for (int j = 0; j < 4; ++j) f.u[4 + j] = *(const unsigned int*)(p + 16 + 2 * j);
  return f.v;
}

// ---------------- GEMM tiling ----------------
constexpr int BM  = 128;   // rows / block
constexpr int BN  = 64;    // cols / panel
constexpr int BK  = 32;    // k-step
constexpr int LDK = 40;    // LDS row stride (ushorts): 80B, 16B-aligned rows

// =====================================================================
// 0) x (fp32) -> bf16
// =====================================================================
__global__ void __launch_bounds__(256)
cvt_x_kernel(const float* __restrict__ x, unsigned short* __restrict__ xb) {
  const int t = blockIdx.x;
  const int c0 = threadIdx.x * 4;
  const float* src = x + (size_t)t * HN + c0;
  unsigned short* dst = xb + (size_t)t * HN + c0;
#pragma unroll
  for (int j = 0; j < 4; ++j) dst[j] = f2bf(src[j]);
}

// =====================================================================
// 1) router: sigmoid scores + grouped top-k (one block / token)
// =====================================================================
__global__ void __launch_bounds__(64)
router_kernel(const float* __restrict__ x, const float* __restrict__ gw,
              const float* __restrict__ ebias,
              int* __restrict__ topk_ids, float* __restrict__ topk_w) {
  const int t = blockIdx.x;
  const int e = threadIdx.x;
  __shared__ float s_sc[EN];
  __shared__ float s_sfc[EN];

  const float* xr = x + (size_t)t * HN;
  const float* wr = gw + (size_t)e * HN;
  float acc = 0.f;
  for (int h = 0; h < HN; h += 4) {
    acc += xr[h] * wr[h] + xr[h + 1] * wr[h + 1] +
           xr[h + 2] * wr[h + 2] + xr[h + 3] * wr[h + 3];
  }
  const float sc = 1.f / (1.f + __expf(-acc));
  s_sc[e] = sc;
  s_sfc[e] = sc + ebias[e];
  __syncthreads();

  if (e == 0) {
    // group score = sum of top-2 corrected scores per group
    float gsc[NGN];
    const int GS = EN / NGN;
    for (int g = 0; g < NGN; ++g) {
      float m1 = -1e30f, m2 = -1e30f;
      for (int j = 0; j < GS; ++j) {
        float v = s_sfc[g * GS + j];
        if (v > m1) { m2 = m1; m1 = v; }
        else if (v > m2) { m2 = v; }
      }
      gsc[g] = m1 + m2;
    }
    // pick TKG groups (ties -> lowest index, matches top_k)
    bool gsel[NGN];
    for (int g = 0; g < NGN; ++g) gsel[g] = false;
    for (int it = 0; it < TKG; ++it) {
      int best = -1; float bv = -1e30f;
      for (int g = 0; g < NGN; ++g)
        if (!gsel[g] && gsc[g] > bv) { bv = gsc[g]; best = g; }
      gsel[best] = true;
    }
    // masked scores, then top-K
    float tmp[EN];
    for (int i = 0; i < EN; ++i) tmp[i] = gsel[i / GS] ? s_sfc[i] : 0.0f;
    int ids[KN];
    float wsum = 0.f;
    for (int kk = 0; kk < KN; ++kk) {
      int best = 0; float bv = -1e30f;
      for (int i = 0; i < EN; ++i)
        if (tmp[i] > bv) { bv = tmp[i]; best = i; }
      tmp[best] = -1e30f;
      ids[kk] = best;
      wsum += s_sc[best];
    }
    const float inv = 1.f / wsum;
    for (int kk = 0; kk < KN; ++kk) {
      topk_ids[t * KN + kk] = ids[kk];
      topk_w[t * KN + kk] = s_sc[ids[kk]] * inv;
    }
  }
}

// =====================================================================
// 2) dispatch build
// =====================================================================
__global__ void zero_kernel(int* __restrict__ cnt, int* __restrict__ cur) {
  if (threadIdx.x < EN) { cnt[threadIdx.x] = 0; cur[threadIdx.x] = 0; }
}
__global__ void hist_kernel(const int* __restrict__ ids, int* __restrict__ cnt) {
  const int i = blockIdx.x * blockDim.x + threadIdx.x;
  if (i < TN * KN) atomicAdd(&cnt[ids[i]], 1);
}
__global__ void scan_kernel(const int* __restrict__ cnt, int* __restrict__ offs) {
  if (threadIdx.x == 0 && blockIdx.x == 0) {
    int s = 0;
    for (int e = 0; e < EN; ++e) { offs[e] = s; s += cnt[e]; }
    offs[EN] = s;
  }
}
__global__ void scatter_kernel(const int* __restrict__ ids,
                               const int* __restrict__ offs, int* __restrict__ cur,
                               int* __restrict__ tok_of_pos,
                               int* __restrict__ assign_pos) {
  const int i = blockIdx.x * blockDim.x + threadIdx.x;
  if (i < TN * KN) {
    const int e = ids[i];
    const int p = offs[e] + atomicAdd(&cur[e], 1);
    tok_of_pos[p] = i / KN;
    assign_pos[i] = p;
  }
}

// =====================================================================
// 3) routed GEMM1: gu = gather(x) @ w1[e], fused SiLU*mul -> h (bf16)
//    grid (E, IN/BN); block 256 = 8 wave32
// =====================================================================
__global__ void __launch_bounds__(256)
routed_gemm1_kernel(const unsigned short* __restrict__ xb,  // [T,H] bf16
                    const float* __restrict__ w1,           // [E,H,2I]
                    const int* __restrict__ tok_of_pos,
                    const int* __restrict__ offs,
                    unsigned short* __restrict__ hout) {    // [T*K, I] bf16
  const int e  = blockIdx.x;
  const int nb = blockIdx.y;                    // gate cols nb*64, up cols IN+nb*64
  const int seg0 = offs[e], seg1 = offs[e + 1];
  const int rows = seg1 - seg0;
  if (rows <= 0) return;

  __shared__ __align__(16) unsigned short As[BM * LDK];
  __shared__ __align__(16) unsigned short Bg[BN * LDK];
  __shared__ __align__(16) unsigned short Bu[BN * LDK];

  const int tid = threadIdx.x;
  const int lane = tid & 31, wave = tid >> 5;
  const int halfsel = lane >> 4, l16 = lane & 15;
  const int arow = tid >> 1, ahalf = tid & 1;   // A stage: 2 threads / row
  const int kk = tid >> 3, n0 = (tid & 7) * 8;  // B stage: 32k x 64n

  const float* w1e = w1 + (size_t)e * HN * (2 * IN);
  unsigned short* amine = As + arow * LDK + ahalf * 16;  // my 32B slice

  for (int mb = 0; mb < rows; mb += BM) {
    int tok = (mb + arow < rows) ? tok_of_pos[seg0 + mb + arow] : -1;
    if (tok < 0) {  // OOB rows stay zero for the whole k loop
      uint4 z = {0, 0, 0, 0};
      ((uint4*)amine)[0] = z; ((uint4*)amine)[1] = z;
    }

    floatx8 accg[4], accu[4];
#pragma unroll
    for (int nt = 0; nt < 4; ++nt) {
#pragma unroll
      for (int r = 0; r < 8; ++r) { accg[nt][r] = 0.f; accu[nt][r] = 0.f; }
    }

    for (int k0 = 0; k0 < HN; k0 += BK) {
      // stage A: async global->LDS (bf16 copy, ASYNCcnt tracked)
      if (tok >= 0) {
        const unsigned short* src = xb + (size_t)tok * HN + k0 + ahalf * 16;
        async_a_b128(src, amine);
        async_a_b128(src + 8, amine + 8);
      }
      // stage B panels, fp32 -> bf16, transposed [n][k]
      {
        const float* gsrc = w1e + (size_t)(k0 + kk) * (2 * IN) + nb * 64 + n0;
        const float* usrc = gsrc + IN;
        if (k0 + BK < HN) __builtin_prefetch(gsrc + (size_t)BK * (2 * IN), 0, 1);
#pragma unroll
        for (int j = 0; j < 8; ++j) {
          Bg[(n0 + j) * LDK + kk] = f2bf(gsrc[j]);
          Bu[(n0 + j) * LDK + kk] = f2bf(usrc[j]);
        }
      }
      async_wait();
      __syncthreads();
      const bf16x16 a = load_frag(As, LDK, wave * 16 + l16, halfsel);
#pragma unroll
      for (int nt = 0; nt < 4; ++nt) {
        const bf16x16 bg = load_frag(Bg, LDK, nt * 16 + l16, halfsel);
        accg[nt] = __builtin_amdgcn_wmma_f32_16x16x32_bf16(
            false, a, false, bg, (short)0, accg[nt], false, false);
        const bf16x16 bu = load_frag(Bu, LDK, nt * 16 + l16, halfsel);
        accu[nt] = __builtin_amdgcn_wmma_f32_16x16x32_bf16(
            false, a, false, bu, (short)0, accu[nt], false, false);
      }
      __syncthreads();
    }
    // epilogue: h = silu(g) * u  (bf16)
#pragma unroll
    for (int nt = 0; nt < 4; ++nt) {
#pragma unroll
      for (int r = 0; r < 8; ++r) {
        const int m = wave * 16 + r + 8 * halfsel;
        if (mb + m < rows) {
          const float g = accg[nt][r], u = accu[nt][r];
          const float s = g / (1.f + __expf(-g));
          const int col = nb * 64 + nt * 16 + l16;
          hout[(size_t)(seg0 + mb + m) * IN + col] = f2bf(s * u);
        }
      }
    }
  }
}

// =====================================================================
// 4) routed GEMM2: ys = h @ w2[e]   grid (E, HN/BN)
// =====================================================================
__global__ void __launch_bounds__(256)
routed_gemm2_kernel(const unsigned short* __restrict__ hin,  // [T*K, I] bf16
                    const float* __restrict__ w2,            // [E, I, H]
                    const int* __restrict__ offs,
                    float* __restrict__ ys) {                // [T*K, H]
  const int e  = blockIdx.x;
  const int nb = blockIdx.y;
  const int seg0 = offs[e], seg1 = offs[e + 1];
  const int rows = seg1 - seg0;
  if (rows <= 0) return;

  __shared__ __align__(16) unsigned short As[BM * LDK];
  __shared__ __align__(16) unsigned short Bs[BN * LDK];

  const int tid = threadIdx.x;
  const int lane = tid & 31, wave = tid >> 5;
  const int halfsel = lane >> 4, l16 = lane & 15;
  const int arow = tid >> 1, ahalf = tid & 1;
  const int kk = tid >> 3, n0 = (tid & 7) * 8;

  const float* w2e = w2 + (size_t)e * IN * HN;
  unsigned short* amine = As + arow * LDK + ahalf * 16;

  for (int mb = 0; mb < rows; mb += BM) {
    const bool valid = (mb + arow < rows);
    if (!valid) {
      uint4 z = {0, 0, 0, 0};
      ((uint4*)amine)[0] = z; ((uint4*)amine)[1] = z;
    }

    floatx8 acc[4];
#pragma unroll
    for (int nt = 0; nt < 4; ++nt)
#pragma unroll
      for (int r = 0; r < 8; ++r) acc[nt][r] = 0.f;

    for (int k0 = 0; k0 < IN; k0 += BK) {
      if (valid) {
        const unsigned short* src =
            hin + (size_t)(seg0 + mb + arow) * IN + k0 + ahalf * 16;
        async_a_b128(src, amine);
        async_a_b128(src + 8, amine + 8);
      }
      {
        const float* bsrc = w2e + (size_t)(k0 + kk) * HN + nb * 64 + n0;
        if (k0 + BK < IN) __builtin_prefetch(bsrc + (size_t)BK * HN, 0, 1);
#pragma unroll
        for (int j = 0; j < 8; ++j) Bs[(n0 + j) * LDK + kk] = f2bf(bsrc[j]);
      }
      async_wait();
      __syncthreads();
      const bf16x16 a = load_frag(As, LDK, wave * 16 + l16, halfsel);
#pragma unroll
      for (int nt = 0; nt < 4; ++nt) {
        const bf16x16 b = load_frag(Bs, LDK, nt * 16 + l16, halfsel);
        acc[nt] = __builtin_amdgcn_wmma_f32_16x16x32_bf16(
            false, a, false, b, (short)0, acc[nt], false, false);
      }
      __syncthreads();
    }
#pragma unroll
    for (int nt = 0; nt < 4; ++nt) {
#pragma unroll
      for (int r = 0; r < 8; ++r) {
        const int m = wave * 16 + r + 8 * halfsel;
        if (mb + m < rows) {
          ys[(size_t)(seg0 + mb + m) * HN + nb * 64 + nt * 16 + l16] = acc[nt][r];
        }
      }
    }
  }
}

// =====================================================================
// 5) shared GEMM1: hs = silu_mul(x @ sw_gu)  grid (T/BM, ISN/BN)
// =====================================================================
__global__ void __launch_bounds__(256)
shared_gemm1_kernel(const unsigned short* __restrict__ xb,  // [T,H] bf16
                    const float* __restrict__ swgu,         // [H, 2*IS]
                    unsigned short* __restrict__ hs) {      // [T, IS] bf16
  const int mb = blockIdx.x * BM;
  const int nb = blockIdx.y;

  __shared__ __align__(16) unsigned short As[BM * LDK];
  __shared__ __align__(16) unsigned short Bg[BN * LDK];
  __shared__ __align__(16) unsigned short Bu[BN * LDK];

  const int tid = threadIdx.x;
  const int lane = tid & 31, wave = tid >> 5;
  const int halfsel = lane >> 4, l16 = lane & 15;
  const int arow = tid >> 1, ahalf = tid & 1;
  const int kk = tid >> 3, n0 = (tid & 7) * 8;

  unsigned short* amine = As + arow * LDK + ahalf * 16;

  floatx8 accg[4], accu[4];
#pragma unroll
  for (int nt = 0; nt < 4; ++nt)
#pragma unroll
    for (int r = 0; r < 8; ++r) { accg[nt][r] = 0.f; accu[nt][r] = 0.f; }

  for (int k0 = 0; k0 < HN; k0 += BK) {
    {
      const unsigned short* src = xb + (size_t)(mb + arow) * HN + k0 + ahalf * 16;
      async_a_b128(src, amine);
      async_a_b128(src + 8, amine + 8);
    }
    {
      const float* gsrc = swgu + (size_t)(k0 + kk) * (2 * ISN) + nb * 64 + n0;
      const float* usrc = gsrc + ISN;
      if (k0 + BK < HN) __builtin_prefetch(gsrc + (size_t)BK * (2 * ISN), 0, 1);
#pragma unroll
      for (int j = 0; j < 8; ++j) {
        Bg[(n0 + j) * LDK + kk] = f2bf(gsrc[j]);
        Bu[(n0 + j) * LDK + kk] = f2bf(usrc[j]);
      }
    }
    async_wait();
    __syncthreads();
    const bf16x16 a = load_frag(As, LDK, wave * 16 + l16, halfsel);
#pragma unroll
    for (int nt = 0; nt < 4; ++nt) {
      const bf16x16 bg = load_frag(Bg, LDK, nt * 16 + l16, halfsel);
      accg[nt] = __builtin_amdgcn_wmma_f32_16x16x32_bf16(
          false, a, false, bg, (short)0, accg[nt], false, false);
      const bf16x16 bu = load_frag(Bu, LDK, nt * 16 + l16, halfsel);
      accu[nt] = __builtin_amdgcn_wmma_f32_16x16x32_bf16(
          false, a, false, bu, (short)0, accu[nt], false, false);
    }
    __syncthreads();
  }
#pragma unroll
  for (int nt = 0; nt < 4; ++nt) {
#pragma unroll
    for (int r = 0; r < 8; ++r) {
      const int m = wave * 16 + r + 8 * halfsel;
      const float g = accg[nt][r], u = accu[nt][r];
      const float s = g / (1.f + __expf(-g));
      hs[(size_t)(mb + m) * ISN + nb * 64 + nt * 16 + l16] = f2bf(s * u);
    }
  }
}

// =====================================================================
// 6) shared GEMM2: out = hs @ sw_d   (full overwrite of d_out)
// =====================================================================
__global__ void __launch_bounds__(256)
shared_gemm2_kernel(const unsigned short* __restrict__ hs,  // [T, IS] bf16
                    const float* __restrict__ swd,          // [IS, H]
                    float* __restrict__ out) {              // [T, H]
  const int mb = blockIdx.x * BM;
  const int nb = blockIdx.y;

  __shared__ __align__(16) unsigned short As[BM * LDK];
  __shared__ __align__(16) unsigned short Bs[BN * LDK];

  const int tid = threadIdx.x;
  const int lane = tid & 31, wave = tid >> 5;
  const int halfsel = lane >> 4, l16 = lane & 15;
  const int arow = tid >> 1, ahalf = tid & 1;
  const int kk = tid >> 3, n0 = (tid & 7) * 8;

  unsigned short* amine = As + arow * LDK + ahalf * 16;

  floatx8 acc[4];
#pragma unroll
  for (int nt = 0; nt < 4; ++nt)
#pragma unroll
    for (int r = 0; r < 8; ++r) acc[nt][r] = 0.f;

  for (int k0 = 0; k0 < ISN; k0 += BK) {
    {
      const unsigned short* src = hs + (size_t)(mb + arow) * ISN + k0 + ahalf * 16;
      async_a_b128(src, amine);
      async_a_b128(src + 8, amine + 8);
    }
    {
      const float* bsrc = swd + (size_t)(k0 + kk) * HN + nb * 64 + n0;
      if (k0 + BK < ISN) __builtin_prefetch(bsrc + (size_t)BK * HN, 0, 1);
#pragma unroll
      for (int j = 0; j < 8; ++j) Bs[(n0 + j) * LDK + kk] = f2bf(bsrc[j]);
    }
    async_wait();
    __syncthreads();
    const bf16x16 a = load_frag(As, LDK, wave * 16 + l16, halfsel);
#pragma unroll
    for (int nt = 0; nt < 4; ++nt) {
      const bf16x16 b = load_frag(Bs, LDK, nt * 16 + l16, halfsel);
      acc[nt] = __builtin_amdgcn_wmma_f32_16x16x32_bf16(
          false, a, false, b, (short)0, acc[nt], false, false);
    }
    __syncthreads();
  }
#pragma unroll
  for (int nt = 0; nt < 4; ++nt) {
#pragma unroll
    for (int r = 0; r < 8; ++r) {
      const int m = wave * 16 + r + 8 * halfsel;
      out[(size_t)(mb + m) * HN + nb * 64 + nt * 16 + l16] = acc[nt][r];
    }
  }
}

// =====================================================================
// 7) combine: out[t] += RSCALE * sum_k w_k * ys[pos(t,k)]  (deterministic)
// =====================================================================
__global__ void __launch_bounds__(256)
combine_kernel(const float* __restrict__ ys, const float* __restrict__ topk_w,
               const int* __restrict__ assign_pos, float* __restrict__ out) {
  const int t = blockIdx.x;
  int pos[KN];
  float w[KN];
#pragma unroll
  for (int kk = 0; kk < KN; ++kk) {
    pos[kk] = assign_pos[t * KN + kk];
    w[kk] = topk_w[t * KN + kk] * RSCALE;
  }
  for (int c = threadIdx.x; c < HN; c += 256) {
    float acc = out[(size_t)t * HN + c];   // shared-expert value
#pragma unroll
    for (int kk = 0; kk < KN; ++kk)
      acc += w[kk] * ys[(size_t)pos[kk] * HN + c];
    out[(size_t)t * HN + c] = acc;
  }
}

// =====================================================================
// launch
// =====================================================================
static inline size_t align256(size_t x) { return (x + 255) & ~(size_t)255; }

extern "C" void kernel_launch(void* const* d_in, const int* in_sizes, int n_in,
                              void* d_out, int out_size, void* d_ws, size_t ws_size,
                              hipStream_t stream) {
  const float* x     = (const float*)d_in[0];   // [T,H]
  const float* gw    = (const float*)d_in[1];   // [E,H]
  const float* ebias = (const float*)d_in[2];   // [E]
  const float* w1    = (const float*)d_in[3];   // [E,H,2I]
  const float* w2    = (const float*)d_in[4];   // [E,I,H]
  const float* swgu  = (const float*)d_in[5];   // [H,2IS]
  const float* swd   = (const float*)d_in[6];   // [IS,H]
  float* out = (float*)d_out;

  // workspace layout
  char* ws = (char*)d_ws;
  size_t o = 0;
  unsigned short* xb = (unsigned short*)(ws + o); o = align256(o + (size_t)TN * HN * 2);
  unsigned short* hs = (unsigned short*)(ws + o); o = align256(o + (size_t)TN * ISN * 2);
  unsigned short* hr = (unsigned short*)(ws + o); o = align256(o + (size_t)TN * KN * IN * 2);
  float* ys          = (float*)(ws + o);          o = align256(o + (size_t)TN * KN * HN * 4);
  int* topk_ids      = (int*)(ws + o);            o = align256(o + (size_t)TN * KN * 4);
  float* topk_w      = (float*)(ws + o);          o = align256(o + (size_t)TN * KN * 4);
  int* tok_of_pos    = (int*)(ws + o);            o = align256(o + (size_t)TN * KN * 4);
  int* assign_pos    = (int*)(ws + o);            o = align256(o + (size_t)TN * KN * 4);
  int* cnt           = (int*)(ws + o);            o = align256(o + (size_t)EN * 4);
  int* offs          = (int*)(ws + o);            o = align256(o + (size_t)(EN + 1) * 4);
  int* cur           = (int*)(ws + o);            o = align256(o + (size_t)EN * 4);
  (void)ws_size; (void)n_in; (void)in_sizes; (void)out_size;

  // 0) activations -> bf16
  cvt_x_kernel<<<TN, 256, 0, stream>>>(x, xb);
  // 1) router
  router_kernel<<<TN, 64, 0, stream>>>(x, gw, ebias, topk_ids, topk_w);
  // 2) dispatch
  zero_kernel<<<1, 64, 0, stream>>>(cnt, cur);
  hist_kernel<<<(TN * KN + 255) / 256, 256, 0, stream>>>(topk_ids, cnt);
  scan_kernel<<<1, 1, 0, stream>>>(cnt, offs);
  scatter_kernel<<<(TN * KN + 255) / 256, 256, 0, stream>>>(
      topk_ids, offs, cur, tok_of_pos, assign_pos);
  // 3) shared expert (writes every element of d_out)
  shared_gemm1_kernel<<<dim3(TN / BM, ISN / BN), 256, 0, stream>>>(xb, swgu, hs);
  shared_gemm2_kernel<<<dim3(TN / BM, HN / BN), 256, 0, stream>>>(hs, swd, out);
  // 4) routed experts
  routed_gemm1_kernel<<<dim3(EN, IN / BN), 256, 0, stream>>>(
      xb, w1, tok_of_pos, offs, hr);
  routed_gemm2_kernel<<<dim3(EN, HN / BN), 256, 0, stream>>>(hr, w2, offs, ys);
  // 5) deterministic combine
  combine_kernel<<<TN, 256, 0, stream>>>(ys, topk_w, assign_pos, out);
}
